// MyNet_386547057362
// MI455X (gfx1250) — matrix operand order repaired and checked by reading, not compiled, not run on hardware
//
#include <hip/hip_runtime.h>
#include <hip/hip_bf16.h>

// ---------------- WMMA fragment types (gfx1250, wave32) ----------------
typedef _Float16 v16h __attribute__((ext_vector_type(16)));
typedef _Float16 h8   __attribute__((ext_vector_type(8)));
typedef float    v8f  __attribute__((ext_vector_type(8)));

typedef unsigned int u32x4 __attribute__((ext_vector_type(4)));
typedef int          i32x4 __attribute__((ext_vector_type(4)));
typedef int          i32x8 __attribute__((ext_vector_type(8)));

#define BNEPS 1e-5f

// TDM builtin: ROCm 7.2 (clang-22) declares a 5-arg form, the upstream
// clang-23 toolchain a 6-arg form (extra zero i32x8). Select by clang major;
// fall back to manual LDS staging when the builtin is absent.
#if __has_builtin(__builtin_amdgcn_tensor_load_to_lds) && \
    __has_builtin(__builtin_amdgcn_s_wait_tensorcnt)
#define USE_TDM 1
#if __clang_major__ >= 23
#define TDM_6ARG 1
#else
#define TDM_6ARG 0
#endif
#else
#define USE_TDM 0
#endif

static __device__ __forceinline__ v16h frag_cat(h8 lo, h8 hi) {
  return __builtin_shufflevector(lo, hi, 0,1,2,3,4,5,6,7,8,9,10,11,12,13,14,15);
}
// A-matrix 16x32 f16: lane<16 holds K {ko..ko+7, ko+16..ko+23} (ko=0), lane>=16 ko=8
static __device__ __forceinline__ v16h load_a_frag(const _Float16* p) {
  return frag_cat(*(const h8*)p, *(const h8*)(p + 16));
}
// B-matrix 32x16 f16: lane<16 holds K {ko..ko+7, ko+16..ko+23} (ko=0), lane>=16 ko=8
static __device__ __forceinline__ v16h load_b_frag(const _Float16* p) {
  return frag_cat(*(const h8*)p, *(const h8*)(p + 16));
}
static __device__ __forceinline__ v8f wmma_f16(v16h a, v16h b, v8f c) {
  return __builtin_amdgcn_wmma_f32_16x16x32_f16(false, a, false, b, (short)0, c, false, false);
}

// ---------------- Stage 1: conv1(3->64,k3) + bias + ReLU + BN -> h1 NHWC f16 [512,30,30,64]
__global__ __launch_bounds__(256)
void k_conv1(const float* __restrict__ x, const float* __restrict__ w,
             const float* __restrict__ bias, const float* __restrict__ g,
             const float* __restrict__ be, const float* __restrict__ mu,
             const float* __restrict__ var, _Float16* __restrict__ h1) {
  int idx = blockIdx.x * 256 + threadIdx.x;     // 512*900*64
  int oc  = idx & 63;
  int pix = idx >> 6;                            // b*900 + y*30 + x
  int xo  = pix % 30; int t = pix / 30;
  int yo  = t % 30;   int b = t / 30;
  const float* xb = x + (size_t)b * 3 * 1024;
  const float* wp = w + oc * 27;
  float acc = 0.f;
#pragma unroll
  for (int ic = 0; ic < 3; ++ic)
#pragma unroll
    for (int ky = 0; ky < 3; ++ky)
#pragma unroll
      for (int kx = 0; kx < 3; ++kx)
        acc += xb[ic * 1024 + (yo + ky) * 32 + (xo + kx)] * wp[ic * 9 + ky * 3 + kx];
  float v = acc + bias[oc];
  v = v > 0.f ? v : 0.f;
  float rs = rsqrtf(var[oc] + BNEPS);
  v = (v - mu[oc]) * rs * g[oc] + be[oc];
  h1[idx] = (_Float16)v;
}

// conv2 weights f32 [256][64][3][3] -> f16 [ky][kx][oc][ic]
__global__ __launch_bounds__(256)
void k_w2cvt(const float* __restrict__ w, _Float16* __restrict__ wh) {
  int idx = blockIdx.x * 256 + threadIdx.x;      // 3*3*256*64
  int ic = idx & 63; int t = idx >> 6;
  int oc = t & 255;  int tap = t >> 8;
  int ky = tap / 3, kx = tap % 3;
  wh[idx] = (_Float16)w[((oc * 64 + ic) * 3 + ky) * 3 + kx];
}

// ---------------- Stage 2: conv2 implicit GEMM, LDS-tiled WMMA ----------------
// Block = 8 waves, output tile 64(M) x 128(N). Per tap: A slice (64x64) and B
// slice (128x64) staged in LDS with rows padded to 72 halves (144B) so the
// 16-lane ds_load_b128 pattern (36*r mod 64) is bank-conflict free.
// B slice moved by the Tensor Data Mover (2D tile, pad writes the padded
// layout directly); manual staging fallback when the builtin is unavailable.
#define LDS_PITCH 72
#define LDSA_H    (64 * LDS_PITCH)     // 4608 halves = 9216 B
#define LDSB_OFF  LDSA_H
#define LDS_BYTES ((64 + 128) * LDS_PITCH * 2)   // 27648 B

__global__ __launch_bounds__(256)
void k_conv2(const _Float16* __restrict__ h1, const _Float16* __restrict__ wh,
             const float* __restrict__ bias, _Float16* __restrict__ out) {
  extern __shared__ _Float16 smem[];
  _Float16* lA = smem;
  _Float16* lB = smem + LDSB_OFF;

  int nblk = blockIdx.x & 1;        // 2 x 128-wide N blocks
  int mblk = blockIdx.x >> 1;       // 6272 x 64-row M blocks
  int m0   = mblk * 64;
  int tid  = threadIdx.x;
  int wv   = tid >> 5;
  int lane = tid & 31;
  int hia  = (lane >> 4) << 3;
  int hib  = (lane >> 4) << 3;

  // A staging role: 4 threads per row, 32B segment each
  int ar = tid >> 2, aseg = (tid & 3) * 16;
  int am = m0 + ar;
  int axo = am % 28; int at = am / 28;
  int ayo = at % 28; int ab = at / 28;
  const _Float16* abase = h1 + ((size_t)(ab * 30 + ayo) * 30 + axo) * 64 + aseg;

  const _Float16* lAr = lA + (lane & 15) * LDS_PITCH;
  const _Float16* lBn = lB + (wv * 16 + (lane & 15)) * LDS_PITCH;

  v8f acc[4] = {};

  for (int tap = 0; tap < 9; ++tap) {
    int ky = tap / 3, kx = tap % 3;
    __syncthreads();                       // previous-tap LDS reads done (WAR)

    // ---- stage A (64 rows x 64 ic) -> padded LDS
    {
      const _Float16* s = abase + (ky * 30 + kx) * 64;
      _Float16* d = lA + ar * LDS_PITCH + aseg;
      *(h8*)d       = *(const h8*)s;
      *(h8*)(d + 8) = *(const h8*)(s + 8);
    }

    // ---- stage B (128 oc x 64 ic) -> padded LDS
#if USE_TDM
    if (wv == 0) {
      unsigned long long va =
          (unsigned long long)(wh + (size_t)(tap * 256 + nblk * 128) * 64);
      u32x4 g0;
      g0[0] = 1u;                                   // count=1, user descriptor
      g0[1] = (unsigned)(size_t)lB;                 // lds_addr (AS3 offset)
      g0[2] = (unsigned)(va & 0xFFFFFFFFu);         // global_addr[31:0]
      g0[3] = (unsigned)((va >> 32) & 0x01FFFFFFu)  // global_addr[56:32]
              | (2u << 30);                         // type=2 (image)
      i32x8 g1;
      g1[0] = (1 << 16)      // data_size = 2B
            | (1 << 20)      // pad_enable
            | (4 << 22)      // pad_interval: 1<<4 * 8B = 128B between pads
            | (3 << 25);     // pad_amount: 4 DWORDs = 16B
      g1[1] = 64 << 16;      // tensor_dim0 = 64 (bits 79:48 low half)
      g1[2] = 128 << 16;     // tensor_dim1 = 128 (bits 111:80 low half)
      g1[3] = 64 << 16;      // tile_dim0 = 64 (bits 127:112)
      g1[4] = 128;           // tile_dim1 = 128 (bits 143:128)
      g1[5] = 64;            // tensor_dim0_stride = 64 elements
      g1[6] = 0;
      g1[7] = 0;
      i32x4 gz = {0, 0, 0, 0};
#if TDM_6ARG
      i32x8 gz8 = {0, 0, 0, 0, 0, 0, 0, 0};
      __builtin_amdgcn_tensor_load_to_lds(g0, g1, gz, gz, gz8, 0);
#else
      __builtin_amdgcn_tensor_load_to_lds(g0, g1, gz, gz, 0);
#endif
      __builtin_amdgcn_s_wait_tensorcnt(0);
    }
#else
    for (int i = tid; i < 1024; i += 256) {        // 1024 x 16B chunks
      int ocl = i >> 3, c8 = (i & 7) * 8;
      const _Float16* s = wh + (size_t)(tap * 256 + nblk * 128 + ocl) * 64 + c8;
      *(h8*)(lB + ocl * LDS_PITCH + c8) = *(const h8*)s;
    }
#endif
    __syncthreads();                       // staged tile visible

    // ---- 8 WMMAs per wave per tap, all operands from LDS
#pragma unroll
    for (int kc = 0; kc < 2; ++kc) {
      v16h bfrag = load_b_frag(lBn + kc * 32 + hib);
#pragma unroll
      for (int mt = 0; mt < 4; ++mt) {
        v16h afrag = load_a_frag(lAr + mt * 16 * LDS_PITCH + kc * 32 + hia);
        acc[mt] = wmma_f16(afrag, bfrag, acc[mt]);
      }
    }
  }

  // ---- epilogue: bias + ReLU, NHWC f16 store
  int n = nblk * 128 + wv * 16 + (lane & 15);
  float bs = bias[n];
#pragma unroll
  for (int mt = 0; mt < 4; ++mt) {
    int rbase = m0 + mt * 16 + ((lane >> 4) << 3);
#pragma unroll
    for (int r = 0; r < 8; ++r) {
      float v = acc[mt][r] + bs;
      v = v > 0.f ? v : 0.f;
      out[(size_t)(rbase + r) * 256 + n] = (_Float16)v;
    }
  }
}

// ---------------- Stage 3: maxpool 2x2 + BN-a -> pooled NHWC f16 [512,14,14,256]
__global__ __launch_bounds__(256)
void k_pool(const _Float16* __restrict__ cin, const float* __restrict__ g,
            const float* __restrict__ be, const float* __restrict__ mu,
            const float* __restrict__ var, _Float16* __restrict__ pooled) {
  int idx = blockIdx.x * 256 + threadIdx.x;      // 512*14*14*256
  int ch = idx & 255; int p = idx >> 8;
  int px = p % 14; int q = p / 14;
  int py = q % 14; int b = q / 14;
  size_t base = ((size_t)(b * 28 + py * 2) * 28 + px * 2) * 256 + ch;
  float v00 = (float)cin[base],            v01 = (float)cin[base + 256];
  float v10 = (float)cin[base + 28 * 256], v11 = (float)cin[base + 28 * 256 + 256];
  float mx = fmaxf(fmaxf(v00, v01), fmaxf(v10, v11));
  float rs = rsqrtf(var[ch] + BNEPS);
  pooled[idx] = (_Float16)((mx - mu[ch]) * rs * g[ch] + be[ch]);
}

// ---------------- Stage 4: grouped conv 64x[4->8,k3] + bias + BN-b -> flat f16 [512][64][1152]
__global__ __launch_bounds__(256)
void k_tconv(const _Float16* __restrict__ pooled, const float* __restrict__ tw,
             const float* __restrict__ tb, const float* __restrict__ g,
             const float* __restrict__ be, const float* __restrict__ mu,
             const float* __restrict__ var, _Float16* __restrict__ flat) {
  int idx = blockIdx.x * 256 + threadIdx.x;      // 512*64*8*144
  int s = idx % 144; int xo = s % 12, yo = s / 12;
  int r = idx / 144;
  int oc8 = r & 7; int bt = r >> 3;
  int t = bt & 63; int b = bt >> 6;
  const float* wp = tw + (size_t)(t * 8 + oc8) * 36;
  const _Float16* pin = pooled + ((size_t)(b * 14 + yo) * 14 + xo) * 256 + t * 4;
  float acc = 0.f;
#pragma unroll
  for (int ic = 0; ic < 4; ++ic)
#pragma unroll
    for (int ky = 0; ky < 3; ++ky)
#pragma unroll
      for (int kx = 0; kx < 3; ++kx)
        acc += (float)pin[(ky * 14 + kx) * 256 + ic] * wp[ic * 9 + ky * 3 + kx];
  int ch = t * 8 + oc8;
  acc += tb[ch];
  float rs = rsqrtf(var[ch] + BNEPS);
  acc = (acc - mu[ch]) * rs * g[ch] + be[ch];
  flat[idx] = (_Float16)acc;   // idx == (b*64+t)*1152 + oc8*144 + s
}

// lin_w f32 [64][9][1152] -> f16 [64][16 padded][1152]
__global__ __launch_bounds__(256)
void k_linwcvt(const float* __restrict__ w, _Float16* __restrict__ wh) {
  int idx = blockIdx.x * 256 + threadIdx.x;      // 64*16*1152
  int k = idx % 1152; int r = idx / 1152;
  int o = r & 15; int t = r >> 4;
  wh[idx] = (o < 9) ? (_Float16)w[(size_t)(t * 9 + o) * 1152 + k] : (_Float16)0.f;
}

// ---------------- Stage 5: 64 tower linears [512x1152]@[1152x9] via WMMA -> z f16 [512][576]
__global__ __launch_bounds__(256)
void k_towgemm(const _Float16* __restrict__ flat, const _Float16* __restrict__ lw,
               const float* __restrict__ lb, _Float16* __restrict__ z) {
  int wave = blockIdx.x * 8 + (threadIdx.x >> 5); // 2048 waves
  int lane = threadIdx.x & 31;
  int t  = wave >> 5;
  int mt = wave & 31;
  const _Float16* ap = flat + ((size_t)(mt * 16 + (lane & 15)) * 64 + t) * 1152;
  const _Float16* bp = lw + ((size_t)t * 16 + (lane & 15)) * 1152;
  int hia = (lane >> 4) << 3;
  int hib = (lane >> 4) << 3;
  v8f c = {};
#pragma unroll 4
  for (int kc = 0; kc < 36; ++kc) {
    v16h a  = load_a_frag(ap + kc * 32 + hia);
    v16h bb = load_b_frag(bp + kc * 32 + hib);
    c = wmma_f16(a, bb, c);
  }
  int o = lane & 15;
  if (o < 9) {
    float bs = lb[t * 9 + o];
    int rbase = mt * 16 + ((lane >> 4) << 3);
#pragma unroll
    for (int r = 0; r < 8; ++r)
      z[(size_t)(rbase + r) * 576 + t * 9 + o] = (_Float16)(c[r] + bs);
  }
}

// fc1_w f32 [512][576] -> f16 (n-major, k contiguous)
__global__ __launch_bounds__(256)
void k_fc1wcvt(const float* __restrict__ w, _Float16* __restrict__ wh) {
  int idx = blockIdx.x * 256 + threadIdx.x;      // 512*576
  wh[idx] = (_Float16)w[idx];
}

// ---------------- Stage 6: fc1 [512x576]@[576x512] + bias + ReLU via WMMA -> h1c f16 [512][512]
__global__ __launch_bounds__(256)
void k_fc1(const _Float16* __restrict__ z, const _Float16* __restrict__ wh,
           const float* __restrict__ bias, _Float16* __restrict__ h1c) {
  int wave = blockIdx.x * 8 + (threadIdx.x >> 5); // 1024 waves
  int lane = threadIdx.x & 31;
  int nt = wave & 31;
  int mt = wave >> 5;
  const _Float16* ap = z  + (size_t)(mt * 16 + (lane & 15)) * 576;
  const _Float16* bp = wh + (size_t)(nt * 16 + (lane & 15)) * 576;
  int hia = (lane >> 4) << 3;
  int hib = (lane >> 4) << 3;
  v8f c = {};
#pragma unroll 3
  for (int kc = 0; kc < 18; ++kc) {
    v16h a  = load_a_frag(ap + kc * 32 + hia);
    v16h bb = load_b_frag(bp + kc * 32 + hib);
    c = wmma_f16(a, bb, c);
  }
  int n = nt * 16 + (lane & 15);
  float bs = bias[n];
  int rbase = mt * 16 + ((lane >> 4) << 3);
#pragma unroll
  for (int r = 0; r < 8; ++r) {
    float v = c[r] + bs;
    v = v > 0.f ? v : 0.f;
    h1c[(size_t)(rbase + r) * 512 + n] = (_Float16)v;
  }
}

// ---------------- Stage 7: fc2 [512x512]@[512x100] + bias -> d_out f32
__global__ __launch_bounds__(256)
void k_fc2(const _Float16* __restrict__ h1c, const float* __restrict__ w,
           const float* __restrict__ bias, float* __restrict__ out) {
  int idx = blockIdx.x * 256 + threadIdx.x;      // 512*100
  int o = idx % 100; int b = idx / 100;
  const _Float16* hp = h1c + (size_t)b * 512;
  const float* wp = w + (size_t)o * 512;
  float acc = bias[o];
#pragma unroll 4
  for (int k = 0; k < 512; ++k)
    acc += (float)hp[k] * wp[k];
  out[idx] = acc;
}

extern "C" void kernel_launch(void* const* d_in, const int* in_sizes, int n_in,
                              void* d_out, int out_size, void* d_ws, size_t ws_size,
                              hipStream_t stream) {
  const float* x       = (const float*)d_in[0];
  const float* conv1_w = (const float*)d_in[1];
  const float* conv1_b = (const float*)d_in[2];
  const float* bn1_g   = (const float*)d_in[3];
  const float* bn1_b   = (const float*)d_in[4];
  const float* bn1_m   = (const float*)d_in[5];
  const float* bn1_v   = (const float*)d_in[6];
  const float* conv2_w = (const float*)d_in[7];
  const float* conv2_b = (const float*)d_in[8];
  const float* bna_g   = (const float*)d_in[9];
  const float* bna_b   = (const float*)d_in[10];
  const float* bna_m   = (const float*)d_in[11];
  const float* bna_v   = (const float*)d_in[12];
  const float* tconv_w = (const float*)d_in[13];
  const float* tconv_b = (const float*)d_in[14];
  const float* bnb_g   = (const float*)d_in[15];
  const float* bnb_b   = (const float*)d_in[16];
  const float* bnb_m   = (const float*)d_in[17];
  const float* bnb_v   = (const float*)d_in[18];
  const float* lin_w   = (const float*)d_in[19];
  const float* lin_b   = (const float*)d_in[20];
  const float* fc1_w   = (const float*)d_in[21];
  const float* fc1_b   = (const float*)d_in[22];
  const float* fc2_w   = (const float*)d_in[23];
  const float* fc2_b   = (const float*)d_in[24];
  float* out = (float*)d_out;

  char* ws = (char*)d_ws;
  _Float16* w2h   = (_Float16*)(ws + 0);          //   294,912 B
  _Float16* linwh = (_Float16*)(ws + 294912);     // 2,359,296 B
  _Float16* fc1wh = (_Float16*)(ws + 2654208);    //   589,824 B
  _Float16* z     = (_Float16*)(ws + 3244032);    //   589,824 B
  _Float16* h1c   = (_Float16*)(ws + 3833856);    //   524,288 B
  _Float16* h1       = (_Float16*)(ws + 4358144);   // 58,982,400 B  [512,30,30,64]
  _Float16* conv2out = (_Float16*)(ws + 63340544);  // 205,520,896 B [512,28,28,256]
  _Float16* pooled   = (_Float16*)(ws + 4358144);   // aliases h1 (dead after conv2)
  _Float16* flat     = (_Float16*)(ws + 63340544);  // aliases conv2out (dead after pool)

  k_w2cvt  <<<576,   256, 0, stream>>>(conv2_w, w2h);
  k_linwcvt<<<4608,  256, 0, stream>>>(lin_w, linwh);
  k_fc1wcvt<<<1152,  256, 0, stream>>>(fc1_w, fc1wh);

  k_conv1<<<115200, 256, 0, stream>>>(x, conv1_w, conv1_b, bn1_g, bn1_b, bn1_m, bn1_v, h1);
  k_conv2<<<12544, 256, LDS_BYTES, stream>>>(h1, w2h, conv2_b, conv2out);
  k_pool<<<100352, 256, 0, stream>>>(conv2out, bna_g, bna_b, bna_m, bna_v, pooled);
  k_tconv<<<147456, 256, 0, stream>>>(pooled, tconv_w, tconv_b, bnb_g, bnb_b, bnb_m, bnb_v, flat);
  k_towgemm<<<256, 256, 0, stream>>>(flat, linwh, lin_b, z);
  k_fc1<<<128, 256, 0, stream>>>(z, fc1wh, fc1_b, h1c);
  k_fc2<<<200, 256, 0, stream>>>(h1c, fc2_w, fc2_b, out);

  (void)in_sizes; (void)n_in; (void)out_size; (void)ws_size;
}
